// MetaSR_74517682585959
// MI455X (gfx1250) — matrix operand order repaired and checked by reading, not compiled, not run on hardware
//
#include <hip/hip_runtime.h>

typedef __attribute__((ext_vector_type(16))) _Float16 v16h;
typedef __attribute__((ext_vector_type(8)))  _Float16 v8h;
typedef __attribute__((ext_vector_type(8)))  float    v8f;

#define QTOT  200000
#define CENC  16
#define VOX   32768   // 32*32*32

// workspace layout
#define FEAT_BYTES ((size_t)CENC * VOX * sizeof(float))   // 2 MB fp32 feature volume
// packed f16 weight fragments (offsets in halfs)
#define W1P 0                    // K=32(pad from 4), N=256 : 1*16*512  = 8192
#define W2P 8192                 // K=256, N=256 : 8*16*512 = 65536
#define W3P (8192 + 65536)
#define W4P (8192 + 2*65536)
#define W5P (8192 + 3*65536)     // K=256, N=432 : 8*27*512 = 110592

// LDS layout (halfs): activations then weight staging buffer.
// Weight buffer = 8 N-tiles x 8 k-chunks = 64 KB so that a block needs only
// 128 KB LDS -> two blocks co-resident per WGP (staging of one block overlaps
// WMMA compute of the other).
#define ACT_HALFS  (8 * 16 * 256)     // 32768 halfs = 64 KB
#define WBUF_HALFS (8 * 8 * 512)      // 32768 halfs = 64 KB

// ---------------------------------------------------------------------------
// Kernel A: 3x3x3 conv, Cin=1 -> 16 channels, SAME padding.
// ---------------------------------------------------------------------------
__global__ void conv3d_kernel(const float* __restrict__ inp,
                              const float* __restrict__ Wenc,
                              const float* __restrict__ benc,
                              float* __restrict__ feat) {
  __shared__ float sw[CENC * 27];
  __shared__ float sb[CENC];
  int tid = threadIdx.x;
  if (tid < CENC * 27) sw[tid] = Wenc[tid];
  if (tid < CENC)      sb[tid] = benc[tid];
  __syncthreads();

  int v = blockIdx.x * blockDim.x + tid;       // 128 blocks * 256 = 32768
  int d = v >> 10, h = (v >> 5) & 31, w = v & 31;

  float acc[CENC];
#pragma unroll
  for (int c = 0; c < CENC; ++c) acc[c] = sb[c];

  for (int i = 0; i < 3; ++i)
    for (int j = 0; j < 3; ++j)
      for (int k = 0; k < 3; ++k) {
        int dd = d + i - 1, hh = h + j - 1, ww = w + k - 1;
        float x = 0.f;
        if ((unsigned)dd < 32u && (unsigned)hh < 32u && (unsigned)ww < 32u)
          x = inp[dd * 1024 + hh * 32 + ww];
        int tap = i * 9 + j * 3 + k;
#pragma unroll
        for (int c = 0; c < CENC; ++c)
          acc[c] = fmaf(sw[c * 27 + tap], x, acc[c]);
      }
#pragma unroll
  for (int c = 0; c < CENC; ++c) feat[c * VOX + v] = acc[c];
}

// ---------------------------------------------------------------------------
// Kernel B: pack fp32 weight matrix W[K][N] into f16 WMMA B-fragments.
//   k = kc*32 + (lane>=16 ? 16 : 0) + h,  n = t*16 + (lane&15)
//   fragment address = (t*nk + kc)*512 + lane*16 + h
// ---------------------------------------------------------------------------
__global__ void pack_w_kernel(const float* __restrict__ W, _Float16* __restrict__ dst,
                              int Kact, int N, int nk, int nt) {
  int total = nk * nt * 512;
  for (int p = blockIdx.x * blockDim.x + threadIdx.x; p < total;
       p += gridDim.x * blockDim.x) {
    int frag = p >> 9, within = p & 511;
    int lane = within >> 4, h = within & 15;
    int kc = frag % nk, t = frag / nk;
    int k = kc * 32 + (lane >> 4) * 16 + h;
    int n = t * 16 + (lane & 15);
    float val = (k < Kact) ? W[k * N + n] : 0.f;
    dst[p] = (_Float16)val;
  }
}

// ---------------------------------------------------------------------------
// Async bulk-stage of packed weights into LDS.
// Each lane issues global_load_async_to_lds_b128 (ASYNCcnt-tracked), 16 B/lane.
// nbytes must be a multiple of 4096 (256 threads * 16 B) -> uniform trip count.
// ---------------------------------------------------------------------------
__device__ __forceinline__ void stage_weights(const _Float16* __restrict__ src,
                                              _Float16* __restrict__ dstLds,
                                              int nbytes) {
  __syncthreads();  // previous consumers of the buffer are done
  uint32_t ldsBase = (uint32_t)(uintptr_t)dstLds;  // low 32 bits = LDS byte offset
  for (int off = threadIdx.x * 16; off < nbytes; off += 256 * 16) {
    uint32_t la = ldsBase + (uint32_t)off;
    asm volatile("global_load_async_to_lds_b128 %0, %1, %2"
                 :: "v"(la), "v"(off), "s"(src)
                 : "memory");
  }
  asm volatile("s_wait_asynccnt 0x0" ::: "memory");
  __syncthreads();  // data visible to all waves
}

// ---------------------------------------------------------------------------
// Fragment helpers
// ---------------------------------------------------------------------------
__device__ __forceinline__ v16h load_a_frag(const _Float16* act, int lane, int kc) {
  int m   = lane & 15;
  int klo = (lane >> 4) * 16;  // lanes 0-15: K 0..15, lanes 16-31: K 16..31
  const _Float16* base = act + m * 256 + kc * 32 + klo;
  v8h lo = *(const v8h*)(base);
  v8h hi = *(const v8h*)(base + 8);
  return __builtin_shufflevector(lo, hi, 0,1,2,3,4,5,6,7,8,9,10,11,12,13,14,15);
}

__device__ __forceinline__ v16h load_b_frag(const _Float16* sW, int frag, int lane) {
  return *(const v16h*)(sW + frag * 512 + lane * 16);  // 32 contiguous bytes/lane
}

__device__ __forceinline__ void store_relu(_Float16* act, const float* __restrict__ bias,
                                           int t, int lane, v8f c) {
  int ln = lane & 15;
  float bv = bias[t * 16 + ln];
  int mbase = (lane >> 4) * 8;
#pragma unroll
  for (int r = 0; r < 8; ++r) {
    float v = fmaxf(c[r] + bv, 0.f);
    act[(mbase + r) * 256 + t * 16 + ln] = (_Float16)v;
  }
}

// ---------------------------------------------------------------------------
// Main fused kernel: 256 threads = 8 waves; each wave owns 16 queries and a
// 16x256 f16 activation slice in LDS. Weights staged per layer-chunk into a
// shared 64 KB LDS buffer via async copies; all waves consume B from LDS.
// ---------------------------------------------------------------------------
__launch_bounds__(256)
__global__ void metasr_main_kernel(const float* __restrict__ coord,
                                   const float* __restrict__ cell,
                                   const float* __restrict__ feat,
                                   const _Float16* __restrict__ pack,
                                   const float* __restrict__ b1,
                                   const float* __restrict__ b2,
                                   const float* __restrict__ b3,
                                   const float* __restrict__ b4,
                                   const float* __restrict__ b5,
                                   float* __restrict__ out) {
  extern __shared__ _Float16 smem[];
  _Float16* sAct = smem;               // 64 KB
  _Float16* sW   = smem + ACT_HALFS;   // 64 KB weight staging
  int tid  = threadIdx.x;
  int wave = tid >> 5;
  int lane = tid & 31;
  _Float16* act = sAct + wave * (16 * 256);

  int qbase = blockIdx.x * 128 + wave * 16;

  // ---- per-query setup: every lane computes for row (lane&15); lanes<16 store x
  int ln = lane & 15;
  int q  = qbase + ln;
  float c0 = 0.f, c1 = 0.f, c2 = 0.f, ce0 = 0.f;
  if (q < QTOT) {
    c0  = coord[q * 3 + 0] - 0.5f * cell[q * 3 + 0];
    c1  = coord[q * 3 + 1] - 0.5f * cell[q * 3 + 1];
    c2  = coord[q * 3 + 2] - 0.5f * cell[q * 3 + 2];
    ce0 = cell[q * 3 + 0];
  }
  const float eps = 1e-6f;
  float cq0 = fminf(fmaxf(c0 + eps, -1.f + eps), 1.f - eps);
  float cq1 = fminf(fmaxf(c1 + eps, -1.f + eps), 1.f - eps);
  float cq2 = fminf(fmaxf(c2 + eps, -1.f + eps), 1.f - eps);
  float i0 = rintf(((cq0 + 1.f) * 32.f - 1.f) * 0.5f);
  float i1 = rintf(((cq1 + 1.f) * 32.f - 1.f) * 0.5f);
  float i2 = rintf(((cq2 + 1.f) * 32.f - 1.f) * 0.5f);
  int validq = (i0 >= 0.f && i0 < 32.f && i1 >= 0.f && i1 < 32.f &&
                i2 >= 0.f && i2 < 32.f && q < QTOT) ? 1 : 0;
  int mdd = min(max((int)i0, 0), 31);
  int mhh = min(max((int)i1, 0), 31);
  int mww = min(max((int)i2, 0), 31);
  float j0 = rintf(((c0 + 1.f) * 32.f - 1.f) * 0.5f);
  float j1 = rintf(((c1 + 1.f) * 32.f - 1.f) * 0.5f);
  float j2 = rintf(((c2 + 1.f) * 32.f - 1.f) * 0.5f);
  int validc = (j0 >= 0.f && j0 < 32.f && j1 >= 0.f && j1 < 32.f &&
                j2 >= 0.f && j2 < 32.f) ? 1 : 0;
  int k0 = min(max((int)j0, 0), 31);
  int k1 = min(max((int)j1, 0), 31);
  int k2 = min(max((int)j2, 0), 31);
  const float inv32 = 1.f / 32.f;
  float adj = (k2 == 0 || k2 == 1 || k2 == 3) ? -inv32 : 0.f;
  float qc0 = -1.f + inv32 + 2.f * inv32 * (float)k0 + adj;
  float qc1 = -1.f + inv32 + 2.f * inv32 * (float)k1 + adj;
  float qc2 = -1.f + inv32 + 2.f * inv32 * (float)k2 + adj;
  if (!validc) { qc0 = 0.f; qc1 = 0.f; qc2 = 0.f; }
  float x0 = (c0 - qc0) * 32.f;
  float x1 = (c1 - qc1) * 32.f;
  float x2 = (c2 - qc2) * 32.f;
  float x3 = ce0 * 16.f;
  if (q >= QTOT) { x0 = x1 = x2 = x3 = 0.f; }

  if (lane < 16) {
    _Float16* row = act + ln * 256;
    row[0] = (_Float16)x0;
    row[1] = (_Float16)x1;
    row[2] = (_Float16)x2;
    row[3] = (_Float16)x3;
#pragma unroll
    for (int z = 4; z < 32; ++z) row[z] = (_Float16)0.f;
  }

  // ---- Layer 1: K=32(pad), N=256, 1 k-chunk (16 fragments = 16 KB, one stage)
  stage_weights(pack + W1P, sW, 16 * 1024);
  {
    v16h a0 = load_a_frag(act, lane, 0);
    for (int t = 0; t < 16; ++t) {
      v8f c = {};
      v16h b = load_b_frag(sW, t, lane);
      c = __builtin_amdgcn_wmma_f32_16x16x32_f16(false, a0, false, b,
                                                 (short)0, c, false, false);
      store_relu(act, b1, t, lane, c);
    }
  }

  // ---- Layers 2..4: K=256, N=256; staged in 2 chunks of 8 N-tiles (64 KB)
  for (int L = 0; L < 3; ++L) {
    const _Float16* wsrc = pack + W2P + (size_t)L * 65536;
    const float* bias = (L == 0) ? b2 : (L == 1) ? b3 : b4;
    v16h a[8];
#pragma unroll
    for (int kc = 0; kc < 8; ++kc) a[kc] = load_a_frag(act, lane, kc);
    for (int t0 = 0; t0 < 16; t0 += 8) {
      stage_weights(wsrc + (size_t)t0 * 8 * 512, sW, 8 * 8 * 1024);
      for (int t = t0; t < t0 + 8; ++t) {
        v8f c = {};
#pragma unroll
        for (int kc = 0; kc < 8; ++kc) {
          v16h b = load_b_frag(sW, (t - t0) * 8 + kc, lane);
          c = __builtin_amdgcn_wmma_f32_16x16x32_f16(false, a[kc], false, b,
                                                     (short)0, c, false, false);
        }
        store_relu(act, bias, t, lane, c);
      }
    }
  }

  // ---- Layer 5 (K=256, N=432) fused with gather-dot epilogue, 8-tile chunks
  {
    v16h a[8];
#pragma unroll
    for (int kc = 0; kc < 8; ++kc) a[kc] = load_a_frag(act, lane, kc);

    int mbase = (lane >> 4) * 8;
    int dd[8], hh[8], ww[8];
    float vq[8];
#pragma unroll
    for (int r = 0; r < 8; ++r) {
      int src = mbase + r;                    // lanes 0..15 hold rows 0..15
      dd[r] = __shfl(mdd, src, 32);
      hh[r] = __shfl(mhh, src, 32);
      ww[r] = __shfl(mww, src, 32);
      vq[r] = (float)__shfl(validq, src, 32);
    }

    float oacc[8];
#pragma unroll
    for (int r = 0; r < 8; ++r) oacc[r] = 0.f;

    for (int t0 = 0; t0 < 27; t0 += 8) {
      int t1 = (t0 + 8 < 27) ? t0 + 8 : 27;
      int nfrag = (t1 - t0) * 8;
      stage_weights(pack + W5P + (size_t)t0 * 8 * 512, sW, nfrag * 1024);

      for (int t = t0; t < t1; ++t) {
        v8f c = {};
#pragma unroll
        for (int kc = 0; kc < 8; ++kc) {
          v16h b = load_b_frag(sW, (t - t0) * 8 + kc, lane);
          c = __builtin_amdgcn_wmma_f32_16x16x32_f16(false, a[kc], false, b,
                                                     (short)0, c, false, false);
        }
        int n  = t * 16 + ln;                 // channel in [0,432)
        float bv = b5[n];
        int cf = n / 27, off = n % 27;
        int di = off / 9 - 1, hj = (off / 3) % 3 - 1, wk = off % 3 - 1;
        const float* fbase = feat + cf * VOX;
#pragma unroll
        for (int r = 0; r < 8; ++r) {
          int d = dd[r] + di, h = hh[r] + hj, w = ww[r] + wk;
          float g = 0.f;
          if ((unsigned)d < 32u && (unsigned)h < 32u && (unsigned)w < 32u)
            g = fbase[d * 1024 + h * 32 + w];
          oacc[r] += (c[r] + bv) * g * vq[r];
        }
      }
    }

    // reduce over the 16 lanes that share each query row
#pragma unroll
    for (int r = 0; r < 8; ++r) {
      float s = oacc[r];
      s += __shfl_xor(s, 1, 32);
      s += __shfl_xor(s, 2, 32);
      s += __shfl_xor(s, 4, 32);
      s += __shfl_xor(s, 8, 32);
      if (ln == 0) {
        int qo = qbase + mbase + r;
        if (qo < QTOT) out[qo] = s;
      }
    }
  }
}

// ---------------------------------------------------------------------------
extern "C" void kernel_launch(void* const* d_in, const int* in_sizes, int n_in,
                              void* d_out, int out_size, void* d_ws, size_t ws_size,
                              hipStream_t stream) {
  const float* inp   = (const float*)d_in[0];
  const float* coord = (const float*)d_in[1];
  const float* cellp = (const float*)d_in[2];
  const float* Wenc  = (const float*)d_in[3];
  const float* benc  = (const float*)d_in[4];
  const float* W1 = (const float*)d_in[5];  const float* b1 = (const float*)d_in[6];
  const float* W2 = (const float*)d_in[7];  const float* b2 = (const float*)d_in[8];
  const float* W3 = (const float*)d_in[9];  const float* b3 = (const float*)d_in[10];
  const float* W4 = (const float*)d_in[11]; const float* b4 = (const float*)d_in[12];
  const float* W5 = (const float*)d_in[13]; const float* b5 = (const float*)d_in[14];
  float* out = (float*)d_out;

  float*     feat = (float*)d_ws;
  _Float16*  pack = (_Float16*)((char*)d_ws + FEAT_BYTES);

  conv3d_kernel<<<VOX / 256, 256, 0, stream>>>(inp, Wenc, benc, feat);

  pack_w_kernel<<< 32, 256, 0, stream>>>(W1, pack + W1P,   4, 256, 1, 16);
  pack_w_kernel<<<256, 256, 0, stream>>>(W2, pack + W2P, 256, 256, 8, 16);
  pack_w_kernel<<<256, 256, 0, stream>>>(W3, pack + W3P, 256, 256, 8, 16);
  pack_w_kernel<<<256, 256, 0, stream>>>(W4, pack + W4P, 256, 256, 8, 16);
  pack_w_kernel<<<256, 256, 0, stream>>>(W5, pack + W5P, 256, 432, 8, 27);

  int blocks = (QTOT + 127) / 128;
  size_t lds = (size_t)(ACT_HALFS + WBUF_HALFS) * sizeof(_Float16);  // 128 KB
  metasr_main_kernel<<<blocks, 256, lds, stream>>>(coord, cellp, feat, pack,
                                                   b1, b2, b3, b4, b5, out);
}
